// Attention_55233279426582
// MI455X (gfx1250) — matrix operand order repaired and checked by reading, not compiled, not run on hardware
//
#include <hip/hip_runtime.h>
#include <hip/hip_bf16.h>

// ---------------------------------------------------------------------------
// MI455X (gfx1250) implementation.
// All conv1x1 / patch-conv / im2col-conv / convT GEMMs and both attention
// matmuls run on v_wmma_f32_16x16x32_f16 (wave32 WMMA). Spatial attention is
// a two-pass streaming softmax that recomputes q^T k with WMMA instead of
// materializing the 268MB attention tensor. All WMMA fragment gathers are
// branch-free LDS reads (zero-pad trick); contiguous A-side gathers are
// packed into 32-bit ds loads.
// ---------------------------------------------------------------------------

typedef _Float16 v16h __attribute__((ext_vector_type(16)));
typedef float    v8f  __attribute__((ext_vector_type(8)));

#define WMMA_F32_F16(A, B, C) \
  __builtin_amdgcn_wmma_f32_16x16x32_f16(false, (A), false, (B), (short)0, (C), false, false)

// A-fragment K index for 16-bit 16x32 A matrix (ISA 7.12.2):
// lane L: row M = L&15, hi = L>>4; element j -> K = (j&7) + (j>>3)*16 + hi*8
// Pairs (2jj, 2jj+1) are K-adjacent -> 32-bit paired gathers are legal.
__device__ __forceinline__ int kmapA(int j, int hi) {
  return (j & 7) + ((j >> 3) << 4) + hi * 8;
}

union frag16 { v16h v; uint u[8]; };

// ---------------------------------------------------------------------------
// Generic WMMA GEMM: OUT[b] = W[MxK] * X[b][KxN]  (per-batch), f16 in, f32 acc
// Block: 256 threads = 8 waves; block tile 16(M) x 128(N); each wave one
// 16x16 output tile per k-chunk. Optional fused per-channel sum/sumsq stats.
// ---------------------------------------------------------------------------
enum { GM_1X1_OI = 0, GM_1X1_IO = 1, GM_PATCH2 = 2, GM_IM2COL3 = 3, GM_CONVT2 = 4 };

template <int MODE>
__global__ void __launch_bounds__(256)
gemm_wmma_k(const float* __restrict__ W, const float* __restrict__ X,
            float* __restrict__ OUT, float* __restrict__ stats,
            int M, int K, int N, int pq)
{
  __shared__ _Float16 As[16 * 32];
  __shared__ _Float16 Bs[32 * 128];
  const int b    = blockIdx.z;
  const int m0   = blockIdx.x * 16;
  const int n0   = blockIdx.y * 128;
  const int tid  = threadIdx.x;
  const int lane = tid & 31;
  const int wave = tid >> 5;
  const int li   = lane & 15;
  const int hi   = lane >> 4;

  constexpr bool PLAIN_B = (MODE != GM_PATCH2 && MODE != GM_IM2COL3);

  v8f acc = {};

  for (int kc = 0; kc < K; kc += 32) {
    // ---- prefetch next B k-chunk (global_prefetch_b8) ----
    if constexpr (PLAIN_B) {
      if (kc + 32 < K) {
        const int kk = tid >> 3, nn = (tid & 7) * 16;
        __builtin_prefetch(&X[((size_t)b * K + kc + 32 + kk) * N + n0 + nn], 0, 1);
      }
    }
    // ---- stage A tile [16 x 32] as f16 ----
    for (int i = tid; i < 16 * 32; i += 256) {
      const int mm = i >> 5, kk = i & 31;
      const int mg = m0 + mm, kg = kc + kk;
      float v;
      if constexpr (MODE == GM_1X1_OI) {
        v = W[(size_t)mg * K + kg];
      } else if constexpr (MODE == GM_1X1_IO) {
        v = W[(size_t)kg * M + mg];
      } else if constexpr (MODE == GM_PATCH2) {       // W[o][c][p][q], k=(c,p,q)
        v = W[(((size_t)mg * (K >> 2) + (kg >> 2)) << 2) + (kg & 3)];
      } else if constexpr (MODE == GM_IM2COL3) {      // W[o][c][3][3], k=(tap,c)
        v = W[((size_t)mg * 64 + (kg & 63)) * 9 + (kg >> 6)];
      } else {                                        // GM_CONVT2: W[i][o][p][q]
        v = W[(((size_t)kg * M + mg) << 2) + pq];
      }
      As[i] = (_Float16)v;
    }
    // ---- stage B tile [32 x 128] as f16 ----
    if constexpr (PLAIN_B) {
      // contiguous in n: float4 global loads, packed 8B LDS stores
      for (int i = tid; i < 32 * 128 / 4; i += 256) {
        const int kk = i >> 5, nn4 = (i & 31) * 4;
        const float4 v4 = *(const float4*)&X[((size_t)b * K + kc + kk) * N + n0 + nn4];
        _Float16 h4[4] = {(_Float16)v4.x, (_Float16)v4.y, (_Float16)v4.z, (_Float16)v4.w};
        *(uint2*)&Bs[kk * 128 + nn4] = *(const uint2*)h4;
      }
    } else {
      for (int i = tid; i < 32 * 128; i += 256) {
        const int kk = i >> 7, nn = i & 127;
        const int kg = kc + kk, ng = n0 + nn;
        float v;
        if constexpr (MODE == GM_PATCH2) {            // 2x2 s2 patches of 64x64 input
          const int c = kg >> 2, p = (kg >> 1) & 1, q = kg & 1;
          const int h = ng >> 5, w = ng & 31;
          v = X[((size_t)b * (K >> 2) + c) * 4096 + (size_t)((h * 2 + p) * 64 + (w * 2 + q))];
        } else {                                      // GM_IM2COL3: 3x3 pad-1 over 32x32
          const int c = kg & 63, tap = kg >> 6;
          const int dy = tap / 3 - 1, dx = tap % 3 - 1;
          const int h = (ng >> 5) + dy, w = (ng & 31) + dx;
          v = (h >= 0 && h < 32 && w >= 0 && w < 32)
                  ? X[((size_t)b * 64 + c) * 1024 + h * 32 + w] : 0.0f;
        }
        Bs[i] = (_Float16)v;
      }
    }
    __syncthreads();

    frag16 af;
    v16h bf;
    #pragma unroll
    for (int jj = 0; jj < 8; ++jj)                    // paired 32-bit gathers
      af.u[jj] = *(const uint*)&As[li * 32 + kmapA(2 * jj, hi)];
    #pragma unroll
    for (int j = 0; j < 16; ++j)
      bf[j] = Bs[kmapA(j, hi) * 128 + wave * 16 + li];   // B: k = kmapA(j, hi)
    acc = WMMA_F32_F16(af.v, bf, acc);
    __syncthreads();
  }

  // ---- store (C/D layout: row = r + hi*8, col = li) + fused BN stats ----
  #pragma unroll
  for (int r = 0; r < 8; ++r) {
    const int mrow = m0 + r + hi * 8;
    const int ncol = n0 + wave * 16 + li;
    const float v = acc[r];
    if constexpr (MODE == GM_CONVT2) {                // scatter to 2x upsampled 64x64
      const int h = ncol >> 5, w = ncol & 31, p = pq >> 1, q = pq & 1;
      OUT[((size_t)b * M + mrow) * 4096 + (size_t)((h * 2 + p) * 64 + (w * 2 + q))] = v;
    } else {
      OUT[((size_t)b * M + mrow) * N + ncol] = v;
    }
    if (stats) {
      float s1 = v, s2 = v * v;
      #pragma unroll
      for (int msk = 1; msk < 16; msk <<= 1) {        // reduce across 16 lanes (wave32)
        s1 += __shfl_xor(s1, msk);
        s2 += __shfl_xor(s2, msk);
      }
      if (li == 0) {
        atomicAdd(&stats[2 * mrow + 0], s1);
        atomicAdd(&stats[2 * mrow + 1], s2);
      }
    }
  }
}

// ---------------------------------------------------------------------------
// BatchNorm(train, gamma=1,beta=0) + ReLU, in place, from accumulated stats
// ---------------------------------------------------------------------------
__global__ void __launch_bounds__(256)
bn_relu_k(float* __restrict__ x, const float* __restrict__ stats,
          int C, int spatial, size_t total, float invN)
{
  const size_t i = (size_t)blockIdx.x * 256 + threadIdx.x;
  if (i >= total) return;
  const int c = (int)((i / spatial) % C);
  const float m  = stats[2 * c] * invN;
  const float vv = fmaxf(stats[2 * c + 1] * invN - m * m, 0.0f);
  const float r  = rsqrtf(vv + 1e-5f);
  const float y  = (x[i] - m) * r;
  x[i] = y > 0.0f ? y : 0.0f;
}

__global__ void zero_k(float* __restrict__ p, int n) {
  const int i = blockIdx.x * 256 + threadIdx.x;
  if (i < n) p[i] = 0.0f;
}

// ---------------------------------------------------------------------------
// Depthwise 3x3 pad-1 over 32x32 (groups = channels), 128 channels
// ---------------------------------------------------------------------------
__global__ void __launch_bounds__(256)
dw3x3_k(const float* __restrict__ in, const float* __restrict__ w, float* __restrict__ out)
{
  const int bc = blockIdx.x;             // b*128 + c
  const int c  = bc & 127;
  __shared__ float wl[9];
  if (threadIdx.x < 9) wl[threadIdx.x] = w[c * 9 + threadIdx.x];
  __syncthreads();
  const size_t base = (size_t)bc * 1024;
  for (int p = threadIdx.x; p < 1024; p += 256) {
    const int h = p >> 5, wd = p & 31;
    float acc = 0.0f;
    #pragma unroll
    for (int dy = -1; dy <= 1; ++dy)
      #pragma unroll
      for (int dx = -1; dx <= 1; ++dx) {
        const int hh = h + dy, ww = wd + dx;
        if (hh >= 0 && hh < 32 && ww >= 0 && ww < 32)
          acc += in[base + hh * 32 + ww] * wl[(dy + 1) * 3 + (dx + 1)];
      }
    out[base + p] = acc;
  }
}

// ---------------------------------------------------------------------------
// L2-normalize a length-1024 row (optional) and pack to f16
// row = b*64 + c ; src channel = chanOff + c inside a chanTot-channel tensor
// ---------------------------------------------------------------------------
__global__ void __launch_bounds__(256)
l2pack_k(const float* __restrict__ in, _Float16* __restrict__ out,
         int chanTot, int chanOff, int normalize)
{
  const int row = blockIdx.x;
  const int b = row >> 6, c = row & 63;
  const float* src = in + ((size_t)b * chanTot + chanOff + c) * 1024;
  _Float16* dst = out + (size_t)row * 1024;
  const int tid = threadIdx.x;
  float vals[4];
  float ss = 0.0f;
  #pragma unroll
  for (int u = 0; u < 4; ++u) { vals[u] = src[tid + u * 256]; ss += vals[u] * vals[u]; }
  #pragma unroll
  for (int msk = 1; msk < 32; msk <<= 1) ss += __shfl_xor(ss, msk);
  __shared__ float part[8];
  __shared__ float scl;
  if ((tid & 31) == 0) part[tid >> 5] = ss;
  __syncthreads();
  if (tid == 0) {
    float t = 0.0f;
    for (int wv = 0; wv < 8; ++wv) t += part[wv];
    scl = normalize ? 1.0f / fmaxf(sqrtf(t), 1e-12f) : 1.0f;
  }
  __syncthreads();
  const float r = scl;
  #pragma unroll
  for (int u = 0; u < 4; ++u) dst[tid + u * 256] = (_Float16)(vals[u] * r);
}

// ---------------------------------------------------------------------------
// Spatial attention pass 1: per-row (n) max and sum-exp of temp * (q^T k).
// q/k staged to LDS (+zero pad); fragments are branch-free ds_load gathers.
// Grid: (bh=64, ntile=64); block 256 = 8 waves, wave covers 128 m columns.
// ---------------------------------------------------------------------------
__global__ void __launch_bounds__(256)
attn_stats_k(const _Float16* __restrict__ qn, const _Float16* __restrict__ kn,
             const float* __restrict__ temp,
             float* __restrict__ rowmax, float* __restrict__ rowsum)
{
  const int bh = blockIdx.x;
  const int b = bh >> 3, h = bh & 7;
  const int nt = blockIdx.y;
  const int tid = threadIdx.x;
  const int lane = tid & 31, wave = tid >> 5;
  const int li = lane & 15, hi = lane >> 4;
  const float tpr = temp[h];
  const size_t base = ((size_t)b * 64 + h * 8) * 1024;

  __shared__ uint kS_u[(8 * 1024 + 16) / 2];   // 8 k rows + zero pad @ 8192
  __shared__ uint qS_u[(8 * 16 + 16) / 2];     // 8x16 q tile + zero pad @ 128
  __shared__ float red[16][8];
  __shared__ float rmx[16];
  _Float16* kS = (_Float16*)kS_u;
  _Float16* qS = (_Float16*)qS_u;

  {
    const uint2* ksrc = (const uint2*)(kn + base);   // 2KB-aligned
    uint2* kdst = (uint2*)kS_u;
    for (int i = tid; i < 2048; i += 256) kdst[i] = ksrc[i];
    if (tid < 128) qS[(tid >> 4) * 16 + (tid & 15)] = qn[base + (size_t)(tid >> 4) * 1024 + nt * 16 + (tid & 15)];
    if (tid < 16) { qS[128 + tid] = (_Float16)0.0f; kS[8 * 1024 + tid] = (_Float16)0.0f; }
  }
  __syncthreads();

  v16h af;
  #pragma unroll
  for (int j = 0; j < 16; ++j) {
    const int c = kmapA(j, hi);
    af[j] = qS[(c < 8 ? c * 16 : 128) + li];
  }

  float mx[8];
  #pragma unroll
  for (int r = 0; r < 8; ++r) mx[r] = -3.0e38f;

  for (int t = 0; t < 8; ++t) {
    const int m0 = wave * 128 + t * 16;
    v16h bf;
    #pragma unroll
    for (int j = 0; j < 16; ++j) {
      const int c = kmapA(j, hi);
      bf[j] = kS[(c < 8 ? c * 1024 + m0 : 8 * 1024) + li];
    }
    v8f z = {};
    v8f s = WMMA_F32_F16(af, bf, z);
    #pragma unroll
    for (int r = 0; r < 8; ++r) mx[r] = fmaxf(mx[r], s[r] * tpr);
  }
  #pragma unroll
  for (int msk = 1; msk < 16; msk <<= 1)
    #pragma unroll
    for (int r = 0; r < 8; ++r) mx[r] = fmaxf(mx[r], __shfl_xor(mx[r], msk));
  if (li == 0)
    #pragma unroll
    for (int r = 0; r < 8; ++r) red[r + hi * 8][wave] = mx[r];
  __syncthreads();
  if (tid < 16) {
    float v = red[tid][0];
    for (int wv = 1; wv < 8; ++wv) v = fmaxf(v, red[tid][wv]);
    rmx[tid] = v;
  }
  __syncthreads();

  float sm[8];
  #pragma unroll
  for (int r = 0; r < 8; ++r) sm[r] = 0.0f;
  for (int t = 0; t < 8; ++t) {
    const int m0 = wave * 128 + t * 16;
    v16h bf;
    #pragma unroll
    for (int j = 0; j < 16; ++j) {
      const int c = kmapA(j, hi);
      bf[j] = kS[(c < 8 ? c * 1024 + m0 : 8 * 1024) + li];
    }
    v8f z = {};
    v8f s = WMMA_F32_F16(af, bf, z);
    #pragma unroll
    for (int r = 0; r < 8; ++r) sm[r] += __expf(s[r] * tpr - rmx[r + hi * 8]);
  }
  #pragma unroll
  for (int msk = 1; msk < 16; msk <<= 1)
    #pragma unroll
    for (int r = 0; r < 8; ++r) sm[r] += __shfl_xor(sm[r], msk);
  if (li == 0)
    #pragma unroll
    for (int r = 0; r < 8; ++r) red[r + hi * 8][wave] = sm[r];
  __syncthreads();
  if (tid < 16) {
    float v = 0.0f;
    for (int wv = 0; wv < 8; ++wv) v += red[tid][wv];
    const int n = nt * 16 + tid;
    rowmax[(size_t)bh * 1024 + n] = rmx[tid];
    rowsum[(size_t)bh * 1024 + n] = v;
  }
}

// ---------------------------------------------------------------------------
// Spatial attention pass 2: out[c,m] = sum_n v[c,n] * softmax_row(n)[m].
// q/v rows and the block's k columns staged in LDS (+zero pads); scores are
// recomputed with WMMA, exponentiated with cached stats, routed through LDS
// into a B-fragment, then WMMA-accumulated as v @ p.
// Grid: (bh=64, 8); block 256 = 8 waves; each wave owns one 16-m tile.
// ---------------------------------------------------------------------------
__global__ void __launch_bounds__(256)
attn_apply_k(const _Float16* __restrict__ qn, const _Float16* __restrict__ kn,
             const _Float16* __restrict__ vh, const float* __restrict__ temp,
             const float* __restrict__ rowmax, const float* __restrict__ rowsum,
             float* __restrict__ outp)
{
  const int bh = blockIdx.x;
  const int b = bh >> 3, h = bh & 7;
  const int tid = threadIdx.x;
  const int lane = tid & 31, wave = tid >> 5;
  const int li = lane & 15, hi = lane >> 4;
  const int mb0 = blockIdx.y * 128;            // block's 128 m columns
  const int m0  = mb0 + wave * 16;             // this wave's 16 m columns
  const float tpr = temp[h];
  const size_t base = ((size_t)b * 64 + h * 8) * 1024;

  __shared__ float smax[1024];
  __shared__ float sinv[1024];
  __shared__ uint qS_u[(8 * 1024 + 16) / 2];   // q rows + zero pad @ 8192
  __shared__ uint vS_u[(8 * 1024 + 16) / 2];   // v rows + zero pad @ 8192
  __shared__ uint kS_u[(8 * 128 + 16) / 2];    // k column block + zero pad @ 1024
  __shared__ _Float16 pl[8][32][16];
  _Float16* qS = (_Float16*)qS_u;
  _Float16* vS = (_Float16*)vS_u;
  _Float16* kS = (_Float16*)kS_u;

  {
    const uint2* qsrc = (const uint2*)(qn + base);
    const uint2* vsrc = (const uint2*)(vh + base);
    uint2* qdst = (uint2*)qS_u;
    uint2* vdst = (uint2*)vS_u;
    for (int i = tid; i < 2048; i += 256) { qdst[i] = qsrc[i]; vdst[i] = vsrc[i]; }
    for (int i = tid; i < 512; i += 256) {     // 8 rows x 128 cols, 2 halves/uint
      const int c = (i * 2) >> 7, col = (i * 2) & 127;
      kS_u[i] = *(const uint*)&kn[base + (size_t)c * 1024 + mb0 + col];
    }
    if (tid < 16) {
      qS[8192 + tid] = (_Float16)0.0f;
      vS[8192 + tid] = (_Float16)0.0f;
      kS[1024 + tid] = (_Float16)0.0f;
    }
    for (int i = tid; i < 1024; i += 256) {
      smax[i] = rowmax[(size_t)bh * 1024 + i];
      sinv[i] = 1.0f / rowsum[(size_t)bh * 1024 + i];
    }
  }
  __syncthreads();

  v16h bk;                                     // k fragment, invariant over n
  #pragma unroll
  for (int j = 0; j < 16; ++j) {
    const int c = kmapA(j, hi);
    bk[j] = kS[(c < 8 ? c * 128 + wave * 16 : 1024) + li];
  }

  v8f acc = {};
  for (int nc = 0; nc < 1024; nc += 32) {
    v16h a0, a1;
    #pragma unroll
    for (int j = 0; j < 16; ++j) {
      const int c = kmapA(j, hi);
      a0[j] = qS[(c < 8 ? c * 1024 + nc : 8192) + li];
      a1[j] = qS[(c < 8 ? c * 1024 + nc + 16 : 8192) + li];
    }
    v8f z0 = {}, z1 = {};
    v8f s0 = WMMA_F32_F16(a0, bk, z0);
    v8f s1 = WMMA_F32_F16(a1, bk, z1);
    #pragma unroll
    for (int r = 0; r < 8; ++r) {
      const int nl = r + hi * 8;
      pl[wave][nl][li]      = (_Float16)(__expf(s0[r] * tpr - smax[nc + nl])      * sinv[nc + nl]);
      pl[wave][16 + nl][li] = (_Float16)(__expf(s1[r] * tpr - smax[nc + 16 + nl]) * sinv[nc + 16 + nl]);
    }
    __syncthreads();
    frag16 av;
    v16h bp;
    #pragma unroll
    for (int jj = 0; jj < 8; ++jj) {           // paired 32-bit gathers (nn contiguous)
      const int nn = kmapA(2 * jj, hi);
      av.u[jj] = *(const uint*)&vS[(li < 8 ? li * 1024 + nc + nn : 8192 + 2 * jj)];
    }
    #pragma unroll
    for (int j = 0; j < 16; ++j)
      bp[j] = pl[wave][kmapA(j, hi)][li];
    acc = WMMA_F32_F16(av.v, bp, acc);
    __syncthreads();
  }
  if (hi == 0) {                               // rows 0..7 hold the 8 real channels
    #pragma unroll
    for (int r = 0; r < 8; ++r)
      outp[base + (size_t)r * 1024 + m0 + li] = acc[r];
  }
}

// ---------------------------------------------------------------------------
// Channel attention (8x8 per (b,h)) — tiny, plain VALU; accumulates into out
// ---------------------------------------------------------------------------
__global__ void __launch_bounds__(256)
chan_attn_k(const _Float16* __restrict__ qn, const _Float16* __restrict__ kn,
            const _Float16* __restrict__ vh, const float* __restrict__ temp,
            float* __restrict__ outp)
{
  const int bh = blockIdx.x;
  const int b = bh >> 3, h = bh & 7;
  const int tid = threadIdx.x;
  const size_t base = ((size_t)b * 64 + h * 8) * 1024;
  __shared__ float P[8][8];
  if (tid < 64) {
    const int c = tid >> 3, d = tid & 7;
    const _Float16* qp = qn + base + (size_t)c * 1024;
    const _Float16* kp = kn + base + (size_t)d * 1024;
    float s = 0.0f;
    for (int n = 0; n < 1024; ++n) s += (float)qp[n] * (float)kp[n];
    P[c][d] = s * temp[h];
  }
  __syncthreads();
  if (tid < 8) {
    float m = -3.0e38f;
    for (int d = 0; d < 8; ++d) m = fmaxf(m, P[tid][d]);
    float s = 0.0f, e[8];
    for (int d = 0; d < 8; ++d) { e[d] = __expf(P[tid][d] - m); s += e[d]; }
    for (int d = 0; d < 8; ++d) P[tid][d] = e[d] / s;
  }
  __syncthreads();
  for (int n = tid; n < 1024; n += 256) {
    float vv[8];
    #pragma unroll
    for (int d = 0; d < 8; ++d) vv[d] = (float)vh[base + (size_t)d * 1024 + n];
    #pragma unroll
    for (int c = 0; c < 8; ++c) {
      float a = 0.0f;
      #pragma unroll
      for (int d = 0; d < 8; ++d) a += P[c][d] * vv[d];
      outp[base + (size_t)c * 1024 + n] += a;
    }
  }
}

// ---------------------------------------------------------------------------
// Host-side orchestration
// ---------------------------------------------------------------------------
extern "C" void kernel_launch(void* const* d_in, const int* in_sizes, int n_in,
                              void* d_out, int out_size, void* d_ws, size_t ws_size,
                              hipStream_t stream)
{
  (void)in_sizes; (void)n_in; (void)out_size; (void)ws_size;
  const float* x      = (const float*)d_in[0];
  const float* y      = (const float*)d_in[1];
  const float* temp   = (const float*)d_in[2];
  const float* enc_w1 = (const float*)d_in[3];
  const float* enc_w2 = (const float*)d_in[4];
  const float* enc_w3 = (const float*)d_in[5];
  const float* kv_w   = (const float*)d_in[6];
  const float* kv_dw  = (const float*)d_in[7];
  const float* q_w    = (const float*)d_in[8];
  const float* q_dw   = (const float*)d_in[9];
  const float* proj_w = (const float*)d_in[10];
  const float* dec_w1 = (const float*)d_in[11];
  const float* dec_w2 = (const float*)d_in[12];
  const float* dec_w3 = (const float*)d_in[13];
  float* out = (float*)d_out;

  char* ws = (char*)d_ws;
  size_t off = 0;
  auto alloc = [&](size_t bytes) -> char* {
    off = (off + 255) & ~(size_t)255;
    char* p = ws + off;
    off += bytes;
    return p;
  };
  float*    ex1 = (float*)alloc(8ull * 32 * 4096 * 4);
  float*    ey1 = (float*)alloc(8ull * 32 * 4096 * 4);
  float*    ex2 = (float*)alloc(8ull * 32 * 1024 * 4);
  float*    ey2 = (float*)alloc(8ull * 32 * 1024 * 4);
  float*    ex3 = (float*)alloc(8ull * 64 * 1024 * 4);
  float*    ey3 = (float*)alloc(8ull * 64 * 1024 * 4);
  float*    kv1 = (float*)alloc(8ull * 128 * 1024 * 4);
  float*    kvd = (float*)alloc(8ull * 128 * 1024 * 4);
  float*    q1  = (float*)alloc(8ull * 64 * 1024 * 4);
  float*    qd  = (float*)alloc(8ull * 64 * 1024 * 4);
  _Float16* qn  = (_Float16*)alloc(8ull * 64 * 1024 * 2);
  _Float16* kn  = (_Float16*)alloc(8ull * 64 * 1024 * 2);
  _Float16* vh  = (_Float16*)alloc(8ull * 64 * 1024 * 2);
  float*    rmx = (float*)alloc(64ull * 1024 * 4);
  float*    rsm = (float*)alloc(64ull * 1024 * 4);
  float*    osc = (float*)alloc(8ull * 64 * 1024 * 4);
  float*    prj = (float*)alloc(8ull * 64 * 1024 * 4);
  float*    dc1 = (float*)alloc(8ull * 128 * 1024 * 4);
  float*    dc2 = (float*)alloc(8ull * 128 * 4096 * 4);
  float*    st  = (float*)alloc(2048 * 4);

  float* st_ex1 = st + 0;     // 32 ch * 2
  float* st_ey1 = st + 64;
  float* st_ex2 = st + 128;
  float* st_ey2 = st + 192;
  float* st_ex3 = st + 256;   // 64 ch * 2
  float* st_ey3 = st + 384;
  float* st_d1  = st + 512;   // 128 ch * 2
  float* st_d2  = st + 768;
  float* st_d3  = st + 1024;  // 256 ch * 2

  const dim3 blk(256);
  zero_k<<<8, blk, 0, stream>>>(st, 2048);

  // ---- encoder(x) ----
  gemm_wmma_k<GM_1X1_OI><<<dim3(2, 32, 8), blk, 0, stream>>>(enc_w1, x, ex1, st_ex1, 32, 256, 4096, 0);
  bn_relu_k<<<4096, blk, 0, stream>>>(ex1, st_ex1, 32, 4096, 8ull * 32 * 4096, 1.0f / 32768.0f);
  gemm_wmma_k<GM_PATCH2><<<dim3(2, 8, 8), blk, 0, stream>>>(enc_w2, ex1, ex2, st_ex2, 32, 128, 1024, 0);
  bn_relu_k<<<1024, blk, 0, stream>>>(ex2, st_ex2, 32, 1024, 8ull * 32 * 1024, 1.0f / 8192.0f);
  gemm_wmma_k<GM_1X1_OI><<<dim3(4, 8, 8), blk, 0, stream>>>(enc_w3, ex2, ex3, st_ex3, 64, 32, 1024, 0);
  bn_relu_k<<<2048, blk, 0, stream>>>(ex3, st_ex3, 64, 1024, 8ull * 64 * 1024, 1.0f / 8192.0f);

  // ---- encoder(y) (shared weights, fresh BN stats) ----
  gemm_wmma_k<GM_1X1_OI><<<dim3(2, 32, 8), blk, 0, stream>>>(enc_w1, y, ey1, st_ey1, 32, 256, 4096, 0);
  bn_relu_k<<<4096, blk, 0, stream>>>(ey1, st_ey1, 32, 4096, 8ull * 32 * 4096, 1.0f / 32768.0f);
  gemm_wmma_k<GM_PATCH2><<<dim3(2, 8, 8), blk, 0, stream>>>(enc_w2, ey1, ey2, st_ey2, 32, 128, 1024, 0);
  bn_relu_k<<<1024, blk, 0, stream>>>(ey2, st_ey2, 32, 1024, 8ull * 32 * 1024, 1.0f / 8192.0f);
  gemm_wmma_k<GM_1X1_OI><<<dim3(4, 8, 8), blk, 0, stream>>>(enc_w3, ey2, ey3, st_ey3, 64, 32, 1024, 0);
  bn_relu_k<<<2048, blk, 0, stream>>>(ey3, st_ey3, 64, 1024, 8ull * 64 * 1024, 1.0f / 8192.0f);

  // ---- kv = dwconv3x3(conv1x1(x_enc)) ; q = conv3x3(conv1x1(y_enc)) ----
  gemm_wmma_k<GM_1X1_OI><<<dim3(8, 8, 8), blk, 0, stream>>>(kv_w, ex3, kv1, nullptr, 128, 64, 1024, 0);
  dw3x3_k<<<1024, blk, 0, stream>>>(kv1, kv_dw, kvd);
  gemm_wmma_k<GM_1X1_OI><<<dim3(4, 8, 8), blk, 0, stream>>>(q_w, ey3, q1, nullptr, 64, 64, 1024, 0);
  gemm_wmma_k<GM_IM2COL3><<<dim3(4, 8, 8), blk, 0, stream>>>(q_dw, q1, qd, nullptr, 64, 576, 1024, 0);

  // ---- l2norm(dim=-1) + f16 pack ----
  l2pack_k<<<512, blk, 0, stream>>>(qd, qn, 64, 0, 1);
  l2pack_k<<<512, blk, 0, stream>>>(kvd, kn, 128, 0, 1);   // k = kvd[:, :64]
  l2pack_k<<<512, blk, 0, stream>>>(kvd, vh, 128, 64, 0);  // v = kvd[:, 64:]

  // ---- spatial attention (streaming softmax, WMMA recompute) ----
  attn_stats_k<<<dim3(64, 64), blk, 0, stream>>>(qn, kn, temp, rmx, rsm);
  attn_apply_k<<<dim3(64, 8), blk, 0, stream>>>(qn, kn, vh, temp, rmx, rsm, osc);
  // ---- channel attention, accumulated into the same buffer ----
  chan_attn_k<<<64, blk, 0, stream>>>(qn, kn, vh, temp, osc);

  // ---- proj (linear => proj(out_s)+proj(out_c) == proj(out_s+out_c)) ----
  gemm_wmma_k<GM_1X1_OI><<<dim3(4, 8, 8), blk, 0, stream>>>(proj_w, osc, prj, nullptr, 64, 64, 1024, 0);

  // ---- decoder ----
  gemm_wmma_k<GM_1X1_IO><<<dim3(8, 8, 8), blk, 0, stream>>>(dec_w1, prj, dc1, st_d1, 128, 64, 1024, 0);
  bn_relu_k<<<4096, blk, 0, stream>>>(dc1, st_d1, 128, 1024, 8ull * 128 * 1024, 1.0f / 8192.0f);
  for (int pq = 0; pq < 4; ++pq)
    gemm_wmma_k<GM_CONVT2><<<dim3(8, 8, 8), blk, 0, stream>>>(dec_w2, dc1, dc2, st_d2, 128, 128, 1024, pq);
  bn_relu_k<<<16384, blk, 0, stream>>>(dc2, st_d2, 128, 4096, 8ull * 128 * 4096, 1.0f / 32768.0f);
  gemm_wmma_k<GM_1X1_IO><<<dim3(16, 32, 8), blk, 0, stream>>>(dec_w3, dc2, out, st_d3, 256, 128, 4096, 0);
  bn_relu_k<<<32768, blk, 0, stream>>>(out, st_d3, 256, 4096, 8ull * 256 * 4096, 1.0f / 32768.0f);
}
